// Transformer_24721831756727
// MI455X (gfx1250) — matrix operand order repaired and checked by reading, not compiled, not run on hardware
//
#include <hip/hip_runtime.h>
#include <math.h>

// ---------------- model constants ----------------
constexpr int VOCAB_ = 32000, D_ = 512, H_ = 8, L_ = 4, DFF_ = 2048, B_ = 8, S_ = 512;
constexpr int DK_ = D_ / H_;        // 64
constexpr int M_ = B_ * S_;         // 4096 rows of activations

typedef __bf16 bf16_t;
typedef __attribute__((ext_vector_type(16))) __bf16 v16bf;
typedef __attribute__((ext_vector_type(8)))  __bf16 v8bf;
typedef __attribute__((ext_vector_type(8)))  float  v8f;

// ---------------- WMMA fragment helpers (wave32; this file's k order, same for A and B) ----------------
// A (16x32 bf16): M = lane%16 ; element e -> K = hi*16+e
__device__ inline v16bf load_frag_a(const bf16_t* __restrict__ base, int ldk,
                                    int m, int k0, int hi) {
  const bf16_t* p = base + (size_t)m * ldk + k0 + hi * 16;
  v8bf a0 = *(const v8bf*)p;
  v8bf a1 = *(const v8bf*)(p + 8);
  v16bf r;
#pragma unroll
  for (int i = 0; i < 8; ++i) { r[i] = a0[i]; r[i + 8] = a1[i]; }
  return r;
}

// B (32x16 bf16) from W stored row-major [N][K]: N = lane%16 ; element e -> K = hi*16+e
__device__ inline v16bf load_frag_b(const bf16_t* __restrict__ base, int ldk,
                                    int n, int k0, int hi) {
  const bf16_t* p = base + (size_t)n * ldk + k0 + hi * 16;
  v8bf b0 = *(const v8bf*)p;
  v8bf b1 = *(const v8bf*)(p + 8);
  v16bf r;
#pragma unroll
  for (int i = 0; i < 8; ++i) { r[i] = b0[i]; r[i + 8] = b1[i]; }
  return r;
}

__device__ inline v8f wmma_bf16(v16bf a, v16bf b, v8f c) {
  return __builtin_amdgcn_wmma_f32_16x16x32_bf16(false, a, false, b,
                                                 (short)0, c, false, false);
}

__device__ inline v8f zero_v8f() {
  v8f z = {0.f, 0.f, 0.f, 0.f, 0.f, 0.f, 0.f, 0.f};
  return z;
}

// async global->LDS copy (CDNA5, tracked by ASYNCcnt). GVS mode: saddr + vaddr32.
__device__ inline void async_copy_b128(const void* saddr, uint32_t goff, uint32_t ldsa) {
  asm volatile("global_load_async_to_lds_b128 %0, %1, %2"
               :: "v"(ldsa), "v"(goff), "s"(saddr) : "memory");
}

// ---------------- kernels ----------------
__global__ __launch_bounds__(256)
void f32_to_bf16_kernel(const float* __restrict__ in, bf16_t* __restrict__ out, size_t n) {
  size_t i = (size_t)blockIdx.x * blockDim.x + threadIdx.x;
  size_t stride = (size_t)gridDim.x * blockDim.x;
  for (; i < n; i += stride) out[i] = (bf16_t)in[i];
}

// x[b*S+s][d] = emb[tok]*scale + PE(s,d)
__global__ __launch_bounds__(256)
void embed_kernel(const int* __restrict__ tok, const float* __restrict__ emb,
                  float* __restrict__ x, int S, int d, float scale, float neg_ln1e4_over_d) {
  int bs = blockIdx.x;
  int s = bs % S;
  int t = tok[bs];
  for (int i = threadIdx.x; i < d; i += 256) {
    float e = (t == 0) ? 0.f : emb[(size_t)t * d + i];
    float div = __expf((float)(2 * (i >> 1)) * neg_ln1e4_over_d);
    float ang = (float)s * div;
    float pe = (i & 1) ? cosf(ang) : sinf(ang);
    x[(size_t)bs * d + i] = e * scale + pe;
  }
}

// row LayerNorm: f32 in -> bf16 out
__global__ __launch_bounds__(256)
void layernorm_kernel(const float* __restrict__ x, const float* __restrict__ g,
                      const float* __restrict__ be, bf16_t* __restrict__ outb, int d) {
  __shared__ float red[256];
  int tid = threadIdx.x;
  const float* p = x + (size_t)blockIdx.x * d;
  float s = 0.f;
  for (int i = tid; i < d; i += 256) s += p[i];
  red[tid] = s; __syncthreads();
  for (int k = 128; k > 0; k >>= 1) { if (tid < k) red[tid] += red[tid + k]; __syncthreads(); }
  float mean = red[0] / (float)d;
  __syncthreads();
  float vs = 0.f;
  for (int i = tid; i < d; i += 256) { float df = p[i] - mean; vs += df * df; }
  red[tid] = vs; __syncthreads();
  for (int k = 128; k > 0; k >>= 1) { if (tid < k) red[tid] += red[tid + k]; __syncthreads(); }
  float rstd = rsqrtf(red[0] / (float)d + 1e-5f);
  bf16_t* o = outb + (size_t)blockIdx.x * d;
  for (int i = tid; i < d; i += 256)
    o[i] = (bf16_t)((p[i] - mean) * rstd * g[i] + be[i]);
}

// C[M,N] = A[M,K](bf16) x W[N,K]^T(bf16) + bias (+residual)(+relu)
// Block: 256 threads = 8 waves, block tile 128M x 64N, wave tile 16M x 64N.
// W tile (64 rows x 32 K = 4KB) double-buffered in LDS via async global->LDS copies,
// shared by all 8 waves; A fragments stream from global (unique rows per wave).
// All fragments for one k-step are loaded into distinct registers before the 4
// back-to-back WMMAs so one coarse wait covers the step (no per-WMMA DS stalls).
__global__ __launch_bounds__(256)
void gemm_lds_kernel(const bf16_t* __restrict__ A, const bf16_t* __restrict__ W,
                     const float* __restrict__ bias, const float* residual,
                     float* Cf, bf16_t* Cb, int M, int N, int K, int relu) {
  __shared__ bf16_t smem[2][64 * 32];           // 2 x 4KB
  int tid = threadIdx.x;
  int wave = tid >> 5, lane = tid & 31;
  int l16 = lane & 15, hi = (lane >> 4) & 1;
  int m0 = blockIdx.y * 128 + wave * 16;
  int n0 = blockIdx.x * 64;

  // cooperative copy mapping: thread -> (row 0..63, 16B chunk 0..3)
  int crow = tid >> 2;
  int cchunk = tid & 3;
  uint32_t lds0 = (uint32_t)(uintptr_t)(&smem[0][0]);
  uint32_t lds_my = lds0 + (uint32_t)(crow * 64 + cchunk * 16);
  uint32_t gbase = (uint32_t)(((size_t)(n0 + crow) * K) * 2 + cchunk * 16);

  v8f acc[4];
#pragma unroll
  for (int t = 0; t < 4; ++t) acc[t] = zero_v8f();

  int nsteps = K >> 5;
  async_copy_b128(W, gbase, lds_my);            // prologue: buffer 0, k0 = 0
  for (int i = 0; i < nsteps; ++i) {
    int k0 = i << 5;
    if (i + 1 < nsteps) {
      async_copy_b128(W, gbase + (uint32_t)((k0 + 32) * 2),
                      lds_my + (uint32_t)(((i + 1) & 1) * 64 * 32 * 2));
      asm volatile("s_wait_asynccnt 0x1" ::: "memory");   // current buffer done
    } else {
      asm volatile("s_wait_asynccnt 0x0" ::: "memory");
    }
    __syncthreads();                             // publish buffer across waves

    const bf16_t* sb = &smem[i & 1][0];
    // phase 1: issue all loads for this k-step into distinct registers
    v16bf a = load_frag_a(A, K, m0 + l16, k0, hi);
    v16bf bfrag[4];
#pragma unroll
    for (int t = 0; t < 4; ++t) {
      const bf16_t* p = sb + (t * 16 + l16) * 32 + hi * 16;
      v8bf b0 = *(const v8bf*)p;
      v8bf b1 = *(const v8bf*)(p + 8);
#pragma unroll
      for (int e = 0; e < 8; ++e) { bfrag[t][e] = b0[e]; bfrag[t][e + 8] = b1[e]; }
    }
    // phase 2: 4 back-to-back WMMAs (independent accumulators, no RAW hazard)
#pragma unroll
    for (int t = 0; t < 4; ++t) acc[t] = wmma_bf16(a, bfrag[t], acc[t]);
    __syncthreads();                             // reads done before buffer reuse
  }

#pragma unroll
  for (int t = 0; t < 4; ++t) {
    int col = n0 + t * 16 + l16;
    float bv = bias ? bias[col] : 0.f;
#pragma unroll
    for (int r = 0; r < 8; ++r) {
      int row = m0 + hi * 8 + r;
      float v = acc[t][r] + bv;
      if (residual) v += residual[(size_t)row * N + col];
      if (relu) v = v > 0.f ? v : 0.f;
      if (Cf) Cf[(size_t)row * N + col] = v;
      if (Cb) Cb[(size_t)row * N + col] = (bf16_t)v;
    }
  }
}

// scores[bh][q][k] = scale * (Q_bh x K_bh^T), with padding/causal mask fused
__global__ __launch_bounds__(32)
void attn_scores_kernel(const bf16_t* __restrict__ Q, const bf16_t* __restrict__ Km,
                        const int* __restrict__ tok, int causal,
                        float* __restrict__ scores, float scale) {
  int bh = blockIdx.z, b = bh / H_, h = bh % H_;
  int lane = threadIdx.x, l16 = lane & 15, hi = lane >> 4;
  int q0 = blockIdx.y * 16, kt0 = blockIdx.x * 16;
  const bf16_t* qbase = Q + (size_t)b * S_ * D_ + h * DK_;
  const bf16_t* kbase = Km + (size_t)b * S_ * D_ + h * DK_;
  v8f acc = zero_v8f();
  for (int kk = 0; kk < DK_; kk += 32) {
    v16bf a = load_frag_a(qbase, D_, q0 + l16, kk, hi);
    v16bf bfr = load_frag_b(kbase, D_, kt0 + l16, kk, hi);
    acc = wmma_bf16(a, bfr, acc);
  }
  int kcol = kt0 + l16;
  int kvalid = tok[b * S_ + kcol] != 0;
#pragma unroll
  for (int r = 0; r < 8; ++r) {
    int qrow = q0 + hi * 8 + r;
    float v = acc[r] * scale;
    int valid = kvalid && (!causal || (kcol <= qrow));
    scores[((size_t)bh * S_ + qrow) * S_ + kcol] = valid ? v : -1e9f;
  }
}

// row softmax over n elements: f32 scores -> bf16 probs
__global__ __launch_bounds__(256)
void softmax_kernel(const float* __restrict__ scores, bf16_t* __restrict__ attn, int n) {
  __shared__ float red[256];
  int tid = threadIdx.x;
  const float* p = scores + (size_t)blockIdx.x * n;
  float mx = -3.4e38f;
  for (int i = tid; i < n; i += 256) mx = fmaxf(mx, p[i]);
  red[tid] = mx; __syncthreads();
  for (int k = 128; k > 0; k >>= 1) { if (tid < k) red[tid] = fmaxf(red[tid], red[tid + k]); __syncthreads(); }
  mx = red[0]; __syncthreads();
  float sum = 0.f;
  for (int i = tid; i < n; i += 256) sum += __expf(p[i] - mx);
  red[tid] = sum; __syncthreads();
  for (int k = 128; k > 0; k >>= 1) { if (tid < k) red[tid] += red[tid + k]; __syncthreads(); }
  float inv = 1.f / red[0];
  bf16_t* o = attn + (size_t)blockIdx.x * n;
  for (int i = tid; i < n; i += 256) o[i] = (bf16_t)(__expf(p[i] - mx) * inv);
}

// ctx[b*S+q][h*DK+d] = attn_bh[q][:] x V_bh[:][d]   (V fragment gathered column-wise)
__global__ __launch_bounds__(32)
void attn_ctx_kernel(const bf16_t* __restrict__ attn, const bf16_t* __restrict__ V,
                     bf16_t* __restrict__ ctx) {
  int bh = blockIdx.z, b = bh / H_, h = bh % H_;
  int lane = threadIdx.x, l16 = lane & 15, hi = lane >> 4;
  int q0 = blockIdx.y * 16, n0 = blockIdx.x * 16;
  const bf16_t* abase = attn + (size_t)bh * S_ * S_;
  const bf16_t* vbase = V + (size_t)b * S_ * D_ + h * DK_;
  int col = n0 + l16;
  v8f acc = zero_v8f();
  for (int k0 = 0; k0 < S_; k0 += 32) {
    v16bf a = load_frag_a(abase, S_, q0 + l16, k0, hi);
    v16bf bb;
#pragma unroll
    for (int e = 0; e < 16; ++e)
      bb[e] = vbase[(size_t)(k0 + hi * 16 + e) * D_ + col];
    acc = wmma_bf16(a, bb, acc);
  }
#pragma unroll
  for (int r = 0; r < 8; ++r)
    ctx[((size_t)(b * S_ + q0 + hi * 8 + r)) * D_ + h * DK_ + col] = (bf16_t)acc[r];
}

// ---------------- host orchestration ----------------
extern "C" void kernel_launch(void* const* d_in, const int* in_sizes, int n_in,
                              void* d_out, int out_size, void* d_ws, size_t ws_size,
                              hipStream_t stream) {
  (void)in_sizes; (void)n_in; (void)out_size; (void)ws_size;
  const int*   src        = (const int*)  d_in[0];
  const int*   tgt        = (const int*)  d_in[1];
  const float* emb        = (const float*)d_in[2];
  const float* enc_attn_w = (const float*)d_in[3];
  const float* enc_attn_b = (const float*)d_in[4];
  const float* enc_ln_g   = (const float*)d_in[5];
  const float* enc_ln_b   = (const float*)d_in[6];
  const float* enc_w1     = (const float*)d_in[7];
  const float* enc_b1     = (const float*)d_in[8];
  const float* enc_w2     = (const float*)d_in[9];
  const float* enc_b2     = (const float*)d_in[10];
  const float* dec_attn_w = (const float*)d_in[11];
  const float* dec_attn_b = (const float*)d_in[12];
  const float* dec_ln_g   = (const float*)d_in[13];
  const float* dec_ln_b   = (const float*)d_in[14];
  const float* dec_w1     = (const float*)d_in[15];
  const float* dec_b1     = (const float*)d_in[16];
  const float* dec_w2     = (const float*)d_in[17];
  const float* dec_b2     = (const float*)d_in[18];
  const float* enc_norm_g = (const float*)d_in[19];
  const float* enc_norm_b = (const float*)d_in[20];
  const float* dec_norm_g = (const float*)d_in[21];
  const float* dec_norm_b = (const float*)d_in[22];

  // carve workspace
  char* wp = (char*)d_ws;
  auto alloc = [&](size_t bytes) -> void* {
    void* p = wp; wp += (bytes + 255) & ~(size_t)255; return p;
  };
  bf16_t* emb_b  = (bf16_t*)alloc((size_t)VOCAB_ * D_ * 2);
  bf16_t* eaw_b  = (bf16_t*)alloc((size_t)L_ * 4 * D_ * D_ * 2);
  bf16_t* daw_b  = (bf16_t*)alloc((size_t)L_ * 8 * D_ * D_ * 2);
  bf16_t* ew1_b  = (bf16_t*)alloc((size_t)L_ * DFF_ * D_ * 2);
  bf16_t* ew2_b  = (bf16_t*)alloc((size_t)L_ * D_ * DFF_ * 2);
  bf16_t* dw1_b  = (bf16_t*)alloc((size_t)L_ * DFF_ * D_ * 2);
  bf16_t* dw2_b  = (bf16_t*)alloc((size_t)L_ * D_ * DFF_ * 2);
  float*  x      = (float*) alloc((size_t)M_ * D_ * 4);
  bf16_t* x2b    = (bf16_t*)alloc((size_t)M_ * D_ * 2);
  bf16_t* qb     = (bf16_t*)alloc((size_t)M_ * D_ * 2);
  bf16_t* kb     = (bf16_t*)alloc((size_t)M_ * D_ * 2);
  bf16_t* vb     = (bf16_t*)alloc((size_t)M_ * D_ * 2);
  float*  scores = (float*) alloc((size_t)B_ * H_ * S_ * S_ * 4);
  bf16_t* attnb  = (bf16_t*)alloc((size_t)B_ * H_ * S_ * S_ * 2);
  bf16_t* ctxb   = (bf16_t*)alloc((size_t)M_ * D_ * 2);
  bf16_t* hb     = (bf16_t*)alloc((size_t)M_ * DFF_ * 2);
  bf16_t* encob  = (bf16_t*)alloc((size_t)M_ * D_ * 2);
  bf16_t* decob  = (bf16_t*)alloc((size_t)M_ * D_ * 2);

  auto conv = [&](const float* in, bf16_t* out, size_t n) {
    int blocks = (int)((n + 255) / 256); if (blocks > 4096) blocks = 4096;
    f32_to_bf16_kernel<<<blocks, 256, 0, stream>>>(in, out, n);
  };
  conv(emb, emb_b, (size_t)VOCAB_ * D_);
  conv(enc_attn_w, eaw_b, (size_t)L_ * 4 * D_ * D_);
  conv(dec_attn_w, daw_b, (size_t)L_ * 8 * D_ * D_);
  conv(enc_w1, ew1_b, (size_t)L_ * DFF_ * D_);
  conv(enc_w2, ew2_b, (size_t)L_ * D_ * DFF_);
  conv(dec_w1, dw1_b, (size_t)L_ * DFF_ * D_);
  conv(dec_w2, dw2_b, (size_t)L_ * D_ * DFF_);

  auto gemm = [&](const bf16_t* A, const bf16_t* W, const float* bias,
                  const float* res, float* Cf, bf16_t* Cb, int N, int K, int relu) {
    dim3 grid(N / 64, M_ / 128);
    gemm_lds_kernel<<<grid, 256, 0, stream>>>(A, W, bias, res, Cf, Cb, M_, N, K, relu);
  };
  auto ln = [&](const float* in, const float* g, const float* b, bf16_t* out) {
    layernorm_kernel<<<M_, 256, 0, stream>>>(in, g, b, out, D_);
  };
  auto attention = [&](const bf16_t* A_in, const bf16_t* KV_in,
                       const bf16_t* Wqkv, const float* Bqkv,  // 3 consecutive [D,D]/[D]
                       const int* tok, int causal) {
    gemm(A_in,  Wqkv + 0 * (size_t)D_ * D_, Bqkv + 0 * D_, nullptr, nullptr, qb, D_, D_, 0);
    gemm(KV_in, Wqkv + 1 * (size_t)D_ * D_, Bqkv + 1 * D_, nullptr, nullptr, kb, D_, D_, 0);
    gemm(KV_in, Wqkv + 2 * (size_t)D_ * D_, Bqkv + 2 * D_, nullptr, nullptr, vb, D_, D_, 0);
    attn_scores_kernel<<<dim3(S_ / 16, S_ / 16, B_ * H_), 32, 0, stream>>>(
        qb, kb, tok, causal, scores, 0.125f);
    softmax_kernel<<<B_ * H_ * S_, 256, 0, stream>>>(scores, attnb, S_);
    attn_ctx_kernel<<<dim3(DK_ / 16, S_ / 16, B_ * H_), 32, 0, stream>>>(attnb, vb, ctxb);
  };

  const float scale = sqrtf((float)D_);
  const float nld = -logf(10000.f) / (float)D_;

  // ---------------- encoder ----------------
  embed_kernel<<<M_, 256, 0, stream>>>(src, emb, x, S_, D_, scale, nld);
  for (int i = 0; i < L_; ++i) {
    const bf16_t* aw = eaw_b + (size_t)i * 4 * D_ * D_;
    const float*  ab = enc_attn_b + (size_t)i * 4 * D_;
    ln(x, enc_ln_g + (size_t)i * 2 * D_, enc_ln_b + (size_t)i * 2 * D_, x2b);
    attention(x2b, x2b, aw, ab, src, 0);
    gemm(ctxb, aw + 3 * (size_t)D_ * D_, ab + 3 * D_, x, x, nullptr, D_, D_, 0);
    ln(x, enc_ln_g + (size_t)i * 2 * D_ + D_, enc_ln_b + (size_t)i * 2 * D_ + D_, x2b);
    gemm(x2b, ew1_b + (size_t)i * DFF_ * D_, enc_b1 + (size_t)i * DFF_,
         nullptr, nullptr, hb, DFF_, D_, 1);
    gemm(hb, ew2_b + (size_t)i * D_ * DFF_, enc_b2 + (size_t)i * D_,
         x, x, nullptr, D_, DFF_, 0);
  }
  layernorm_kernel<<<M_, 256, 0, stream>>>(x, enc_norm_g, enc_norm_b, encob, D_);

  // ---------------- decoder ----------------
  embed_kernel<<<M_, 256, 0, stream>>>(tgt, emb, x, S_, D_, scale, nld);
  for (int i = 0; i < L_; ++i) {
    const bf16_t* aw = daw_b + (size_t)i * 8 * D_ * D_;
    const float*  ab = dec_attn_b + (size_t)i * 8 * D_;
    const float*  lg = dec_ln_g + (size_t)i * 3 * D_;
    const float*  lb = dec_ln_b + (size_t)i * 3 * D_;
    // masked self-attention
    ln(x, lg, lb, x2b);
    attention(x2b, x2b, aw, ab, tgt, 1);
    gemm(ctxb, aw + 3 * (size_t)D_ * D_, ab + 3 * D_, x, x, nullptr, D_, D_, 0);
    // cross-attention (K/V from encoder output)
    ln(x, lg + D_, lb + D_, x2b);
    attention(x2b, encob, aw + 4 * (size_t)D_ * D_, ab + 4 * D_, src, 0);
    gemm(ctxb, aw + 7 * (size_t)D_ * D_, ab + 7 * D_, x, x, nullptr, D_, D_, 0);
    // FFN
    ln(x, lg + 2 * D_, lb + 2 * D_, x2b);
    gemm(x2b, dw1_b + (size_t)i * DFF_ * D_, dec_b1 + (size_t)i * DFF_,
         nullptr, nullptr, hb, DFF_, D_, 1);
    gemm(hb, dw2_b + (size_t)i * D_ * DFF_, dec_b2 + (size_t)i * D_,
         x, x, nullptr, D_, DFF_, 0);
  }
  layernorm_kernel<<<M_, 256, 0, stream>>>(x, dec_norm_g, dec_norm_b, decob, D_);

  // tied output projection: logits = dec_out x emb^T  -> d_out (f32)
  gemm(decob, emb_b, nullptr, nullptr, (float*)d_out, nullptr, VOCAB_, D_, 0);
}
